// Saliency_Mining_Module_28544352649854
// MI455X (gfx1250) — matrix operand order repaired and checked by reading, not compiled, not run on hardware
//
#include <hip/hip_runtime.h>
#include <hip/hip_bf16.h>

typedef __attribute__((ext_vector_type(16))) _Float16 v16h;
typedef __attribute__((ext_vector_type(8)))  _Float16 v8h;
typedef __attribute__((ext_vector_type(2)))  _Float16 v2h;
typedef __attribute__((ext_vector_type(8)))  float    v8f;

#define N_IMG 8
#define C1N   64
#define H     160
#define W     160
#define HW    (H * W)            // 25600
#define CHW   (C1N * HW)         // 1638400
#define TOT   (N_IMG * CHW)      // 13107200
#define HH    80
#define HWH   (HH * HH)          // 6400
#define EPS   1e-5f

static __device__ __forceinline__ int imin(int a, int b) { return a < b ? a : b; }
static __device__ __forceinline__ int imax(int a, int b) { return a > b ? a : b; }

// ---------------------------------------------------------------------------
// 1) bilinear x2 upsample (align_corners=True) of high_stage: [8,64,80,80]->[8,64,160,160]
// ---------------------------------------------------------------------------
__global__ void k_resize_high(const float* __restrict__ hs, float* __restrict__ out) {
    int idx = blockIdx.x * blockDim.x + threadIdx.x;
    if (idx >= TOT) return;
    int x  = idx % W;
    int t  = idx / W;
    int y  = t % H;
    int nc = t / H;
    const float scale = 79.0f / 159.0f;   // (in-1)/(out-1)
    float py = y * scale, px = x * scale;
    int y0 = (int)py, x0 = (int)px;       // nonneg -> trunc == floor
    float fy = py - y0, fx = px - x0;
    int y1 = imin(y0 + 1, HH - 1), x1 = imin(x0 + 1, HH - 1);
    const float* p = hs + (long)nc * HWH;
    float v = (1.f - fy) * ((1.f - fx) * p[y0 * HH + x0] + fx * p[y0 * HH + x1])
            +         fy * ((1.f - fx) * p[y1 * HH + x0] + fx * p[y1 * HH + x1]);
    out[idx] = v;
}

// ---------------------------------------------------------------------------
// 2) sigmoid(resize(in_map)): [8,1,80,80] -> [8,160,160]
// ---------------------------------------------------------------------------
__global__ void k_imap(const float* __restrict__ im, float* __restrict__ out) {
    int idx = blockIdx.x * blockDim.x + threadIdx.x;
    if (idx >= N_IMG * HW) return;
    int x = idx % W;
    int t = idx / W;
    int y = t % H;
    int n = t / H;
    const float scale = 79.0f / 159.0f;
    float py = y * scale, px = x * scale;
    int y0 = (int)py, x0 = (int)px;
    float fy = py - y0, fx = px - x0;
    int y1 = imin(y0 + 1, HH - 1), x1 = imin(x0 + 1, HH - 1);
    const float* p = im + (long)n * HWH;
    float v = (1.f - fy) * ((1.f - fx) * p[y0 * HH + x0] + fx * p[y0 * HH + x1])
            +         fy * ((1.f - fx) * p[y1 * HH + x0] + fx * p[y1 * HH + x1]);
    out[idx] = 1.0f / (1.0f + __expf(-v));
}

// ---------------------------------------------------------------------------
// 3) Dual 1x1-conv GEMM + fused BN via WMMA (f16 in, f32 acc).
//    out[pix, oc] = sum_k cat[pix, k] * w1[oc, k];  cat = [low(64) ; high(64)]
//    Block: 64-pixel M tile, K=128, N=64; BOTH weight sets share the A tile.
//    Waves 0-3 -> dg1/t1, waves 4-7 -> dg2/t2; each wave: 16 v_wmma ops.
//    B is pre-swizzled fragment-major in LDS -> each fragment = 2x ds_load_b128.
//    A fragments = 2x aligned v8h LDS loads (lane half h: contiguous k = 16h..16h+15, as in B).
// ---------------------------------------------------------------------------
__global__ __launch_bounds__(256)
void k_dual_gemm_bn(const float* __restrict__ low, const float* __restrict__ high,
                    const float* __restrict__ w1a, const float* __restrict__ w1b,
                    const float* __restrict__ sc1, const float* __restrict__ bi1,
                    const float* __restrict__ mn1, const float* __restrict__ vr1,
                    const float* __restrict__ sc2, const float* __restrict__ bi2,
                    const float* __restrict__ mn2, const float* __restrict__ vr2,
                    _Float16* __restrict__ t1, _Float16* __restrict__ t2) {
    __shared__ __align__(32) _Float16 As[64][128];        // [pixel][k]  16 KB
    __shared__ __align__(32) v2h Bf[2][4][4][32][8];      // [set][ks][nt][lane][j2] 32 KB

    const int tid = threadIdx.x;
    const long pixBase = (long)blockIdx.x * 64;

    // ---- stage A: 64 pixels x 128 channels; float2 global loads (coalesced)
    #pragma unroll
    for (int it = 0; it < 16; ++it) {
        int e = it * 256 + tid;           // 0..4095
        int p = (e & 31) * 2;
        int k = e >> 5;                   // 0..127
        long pix = pixBase + p;
        int n   = (int)(pix / HW);
        int loc = (int)(pix % HW);
        const float* src = (k < C1N)
                ? (low  + (long)n * CHW + (long)k         * HW + loc)
                : (high + (long)n * CHW + (long)(k - C1N) * HW + loc);
        float2 v = *(const float2*)src;
        As[p][k]     = (_Float16)v.x;
        As[p + 1][k] = (_Float16)v.y;
    }
    // ---- stage B: both sets, directly into fragment-major (WMMA B) layout
    #pragma unroll
    for (int it = 0; it < 16; ++it) {
        int e    = it * 256 + tid;        // 0..4095
        int j2   = e & 7;
        int lane = (e >> 3) & 31;
        int nt   = (e >> 8) & 3;
        int ks   = (e >> 10) & 3;
        int o = nt * 16 + (lane & 15);
        int k = ks * 32 + ((lane >= 16) ? 16 : 0) + j2 * 2;
        float2 wa = *(const float2*)(w1a + o * 128 + k);
        float2 wb = *(const float2*)(w1b + o * 128 + k);
        v2h pa; pa[0] = (_Float16)wa.x; pa[1] = (_Float16)wa.y;
        v2h pb; pb[0] = (_Float16)wb.x; pb[1] = (_Float16)wb.y;
        Bf[0][ks][nt][lane][j2] = pa;
        Bf[1][ks][nt][lane][j2] = pb;
    }
    __syncthreads();

    const int wave = tid >> 5;
    const int lane = tid & 31;
    const int ws   = wave >> 2;           // 0 -> dg1, 1 -> dg2
    const int mt   = wave & 3;            // M subtile (16 rows)
    const int lo16 = lane & 15;
    const int hiHalf = (lane >= 16) ? 1 : 0;
    const int mrow = mt * 16 + lo16;

    v8f acc[4] = {};                       // 4 N-subtiles of 16

    #pragma unroll
    for (int ksIdx = 0; ksIdx < 4; ++ksIdx) {
        const int kbase = ksIdx * 32;
        // A fragment: lane holds row mrow; two contiguous 8-half runs
        const _Float16* arow = &As[mrow][0];
        v8h alo = *(const v8h*)(arow + kbase + 16 * hiHalf);
        v8h ahi = *(const v8h*)(arow + kbase + 16 * hiHalf + 8);
        v16h a = __builtin_shufflevector(alo, ahi,
                                         0, 1, 2, 3, 4, 5, 6, 7,
                                         8, 9, 10, 11, 12, 13, 14, 15);
        #pragma unroll
        for (int nt = 0; nt < 4; ++nt) {
            v16h b = *(const v16h*)&Bf[ws][ksIdx][nt][lane][0];   // 32B contiguous
            acc[nt] = __builtin_amdgcn_wmma_f32_16x16x32_f16(
                false, a, false, b, (short)0, acc[nt], false, false);
        }
    }

    // ---- epilogue: BN fused, one v8h (16B) store per N-subtile
    const float* sc = ws ? sc2 : sc1;
    const float* bi = ws ? bi2 : bi1;
    const float* mn = ws ? mn2 : mn1;
    const float* vr = ws ? vr2 : vr1;
    _Float16*    to = ws ? t2  : t1;

    const long pixRow = pixBase + mt * 16 + hiHalf * 8;  // 8 consecutive pixels
    const int ni  = (int)(pixRow / HW);
    const int loc = (int)(pixRow % HW);

    #pragma unroll
    for (int nt = 0; nt < 4; ++nt) {
        const int n = nt * 16 + lo16;
        const float inv = sc[n] * rsqrtf(vr[n] + EPS);
        const float b2  = bi[n] - mn[n] * inv;
        v8h hh;
        #pragma unroll
        for (int r = 0; r < 8; ++r) hh[r] = (_Float16)(acc[nt][r] * inv + b2);
        *(v8h*)(to + (long)ni * CHW + (long)n * HW + loc) = hh;
    }
}

// ---------------------------------------------------------------------------
// 4) 3x3 conv 64 -> 2 (zero pad), f16 input, produces flow field d [8,2,160,160]
// ---------------------------------------------------------------------------
__global__ void k_conv3x3_64to2(const _Float16* __restrict__ t, const float* __restrict__ w2,
                                float* __restrict__ d) {
    int idx = blockIdx.x * blockDim.x + threadIdx.x;
    if (idx >= N_IMG * HW) return;
    int x = idx % W;
    int q = idx / W;
    int y = q % H;
    int n = q / H;
    float a0 = 0.f, a1 = 0.f;
    const _Float16* base = t + (long)n * CHW;
    for (int c = 0; c < C1N; ++c) {
        const _Float16* pl  = base + (long)c * HW;
        const float*    wc0 = w2 + (0 * C1N + c) * 9;
        const float*    wc1 = w2 + (1 * C1N + c) * 9;
        #pragma unroll
        for (int dy = -1; dy <= 1; ++dy) {
            int yy = y + dy;
            if ((unsigned)yy >= (unsigned)H) continue;
            #pragma unroll
            for (int dx = -1; dx <= 1; ++dx) {
                int xx = x + dx;
                if ((unsigned)xx >= (unsigned)W) continue;
                float v = (float)pl[yy * W + xx];
                int wi = (dy + 1) * 3 + (dx + 1);
                a0 = fmaf(v, wc0[wi], a0);
                a1 = fmaf(v, wc1[wi], a1);
            }
        }
    }
    d[(long)n * 2 * HW + y * W + x]      = a0;
    d[(long)n * 2 * HW + HW + y * W + x] = a1;
}

// ---------------------------------------------------------------------------
// 5) Fused: hi = gridsample(high, base+d1/160); lo = gridsample(low, base+d2/160);
//    hs = hi + lo;  f = gamma*hs*imap + hs   (f -> d_out)
// ---------------------------------------------------------------------------
__device__ __forceinline__ void mk_gather(float px, float py, int idx[4], float wgt[4]) {
    float x0f = floorf(px), y0f = floorf(py);
    float fx = px - x0f, fy = py - y0f;
    int x0 = (int)x0f, y0 = (int)y0f;
    int x1 = x0 + 1, y1 = y0 + 1;
    float vx0 = (x0 >= 0 && x0 < W) ? 1.f : 0.f;
    float vx1 = (x1 >= 0 && x1 < W) ? 1.f : 0.f;
    float vy0 = (y0 >= 0 && y0 < H) ? 1.f : 0.f;
    float vy1 = (y1 >= 0 && y1 < H) ? 1.f : 0.f;
    int xc0 = imin(imax(x0, 0), W - 1), xc1 = imin(imax(x1, 0), W - 1);
    int yc0 = imin(imax(y0, 0), H - 1), yc1 = imin(imax(y1, 0), H - 1);
    idx[0] = yc0 * W + xc0; wgt[0] = (1.f - fy) * (1.f - fx) * vy0 * vx0;
    idx[1] = yc0 * W + xc1; wgt[1] = (1.f - fy) * fx * vy0 * vx1;
    idx[2] = yc1 * W + xc0; wgt[2] = fy * (1.f - fx) * vy1 * vx0;
    idx[3] = yc1 * W + xc1; wgt[3] = fy * fx * vy1 * vx1;
}

__global__ void k_warp_fuse(const float* __restrict__ low, const float* __restrict__ high,
                            const float* __restrict__ d1, const float* __restrict__ d2,
                            const float* __restrict__ imap, const float* __restrict__ gamma,
                            float* __restrict__ f) {
    int idx = blockIdx.x * blockDim.x + threadIdx.x;
    if (idx >= N_IMG * HW) return;
    int x = idx % W;
    int q = idx / W;
    int y = q % H;
    int n = q / H;

    const float gx = -1.f + 2.f * (float)x / 159.f;
    const float gy = -1.f + 2.f * (float)y / 159.f;
    const long dof = (long)n * 2 * HW + y * W + x;
    const float inv160 = 1.f / 160.f;

    float px1 = (gx + d1[dof] * inv160 + 1.f) * 80.f - 0.5f;
    float py1 = (gy + d1[dof + HW] * inv160 + 1.f) * 80.f - 0.5f;
    float px2 = (gx + d2[dof] * inv160 + 1.f) * 80.f - 0.5f;
    float py2 = (gy + d2[dof + HW] * inv160 + 1.f) * 80.f - 0.5f;

    int ih[4], il[4];
    float wh[4], wl[4];
    mk_gather(px1, py1, ih, wh);
    mk_gather(px2, py2, il, wl);

    const float g  = gamma[0];
    const float im = imap[(long)n * HW + y * W + x];
    const float s  = g * im + 1.f;        // f = hs*(gamma*imap + 1)

    const float* hb = high + (long)n * CHW;
    const float* lb = low  + (long)n * CHW;
    float* fb = f + (long)n * CHW + y * W + x;

    for (int c = 0; c < C1N; ++c) {
        const float* hp = hb + (long)c * HW;
        const float* lp = lb + (long)c * HW;
        float hv = hp[ih[0]] * wh[0] + hp[ih[1]] * wh[1] + hp[ih[2]] * wh[2] + hp[ih[3]] * wh[3];
        float lv = lp[il[0]] * wl[0] + lp[il[1]] * wl[1] + lp[il[2]] * wl[2] + lp[il[3]] * wl[3];
        fb[(long)c * HW] = (hv + lv) * s;
    }
}

// ---------------------------------------------------------------------------
// 6) out conv: 3x3, 64 -> 1, + bias; reads f from d_out, writes o after it
// ---------------------------------------------------------------------------
__global__ void k_outconv(const float* __restrict__ f, const float* __restrict__ wout,
                          const float* __restrict__ bout, float* __restrict__ o) {
    int idx = blockIdx.x * blockDim.x + threadIdx.x;
    if (idx >= N_IMG * HW) return;
    int x = idx % W;
    int q = idx / W;
    int y = q % H;
    int n = q / H;
    float acc = bout[0];
    const float* base = f + (long)n * CHW;
    for (int c = 0; c < C1N; ++c) {
        const float* pl = base + (long)c * HW;
        const float* wc = wout + c * 9;
        #pragma unroll
        for (int dy = -1; dy <= 1; ++dy) {
            int yy = y + dy;
            if ((unsigned)yy >= (unsigned)H) continue;
            #pragma unroll
            for (int dx = -1; dx <= 1; ++dx) {
                int xx = x + dx;
                if ((unsigned)xx >= (unsigned)W) continue;
                acc = fmaf(pl[yy * W + xx], wc[(dy + 1) * 3 + (dx + 1)], acc);
            }
        }
    }
    o[(long)n * HW + y * W + x] = acc;
}

// ---------------------------------------------------------------------------
// workspace layout (bytes)
// ---------------------------------------------------------------------------
static constexpr size_t OFF_HIGH = 0;                                  // f32 [8,64,160,160]
static constexpr size_t OFF_IMAP = OFF_HIGH + (size_t)TOT * 4;         // f32 [8,160,160]
static constexpr size_t OFF_D1   = OFF_IMAP + (size_t)N_IMG * HW * 4;  // f32 [8,2,160,160]
static constexpr size_t OFF_D2   = OFF_D1 + (size_t)N_IMG * 2 * HW * 4;
static constexpr size_t OFF_T1   = OFF_D2 + (size_t)N_IMG * 2 * HW * 4; // f16 [8,64,160,160]
static constexpr size_t OFF_T2   = OFF_T1 + (size_t)TOT * 2;
// total ~104 MB

extern "C" void kernel_launch(void* const* d_in, const int* in_sizes, int n_in,
                              void* d_out, int out_size, void* d_ws, size_t ws_size,
                              hipStream_t stream) {
    const float* low    = (const float*)d_in[0];
    const float* hstage = (const float*)d_in[1];
    const float* inmap  = (const float*)d_in[2];
    const float* dg1_w1 = (const float*)d_in[3];
    const float* dg1_sc = (const float*)d_in[4];
    const float* dg1_bi = (const float*)d_in[5];
    const float* dg1_mn = (const float*)d_in[6];
    const float* dg1_vr = (const float*)d_in[7];
    const float* dg1_w2 = (const float*)d_in[8];
    const float* dg2_w1 = (const float*)d_in[9];
    const float* dg2_sc = (const float*)d_in[10];
    const float* dg2_bi = (const float*)d_in[11];
    const float* dg2_mn = (const float*)d_in[12];
    const float* dg2_vr = (const float*)d_in[13];
    const float* dg2_w2 = (const float*)d_in[14];
    const float* gamma  = (const float*)d_in[15];
    const float* out_w  = (const float*)d_in[16];
    const float* out_b  = (const float*)d_in[17];

    char* ws = (char*)d_ws;
    float*    high = (float*)(ws + OFF_HIGH);
    float*    imap = (float*)(ws + OFF_IMAP);
    float*    d1   = (float*)(ws + OFF_D1);
    float*    d2   = (float*)(ws + OFF_D2);
    _Float16* t1   = (_Float16*)(ws + OFF_T1);
    _Float16* t2   = (_Float16*)(ws + OFF_T2);

    float* f = (float*)d_out;       // [8,64,160,160]
    float* o = f + TOT;             // [8,1,160,160]

    k_resize_high<<<TOT / 256, 256, 0, stream>>>(hstage, high);
    k_imap<<<(N_IMG * HW) / 256, 256, 0, stream>>>(inmap, imap);

    // 204800 pixels / 64-pixel tiles = 3200 blocks; both GEMMs share the A tile
    k_dual_gemm_bn<<<3200, 256, 0, stream>>>(low, high, dg1_w1, dg2_w1,
                                             dg1_sc, dg1_bi, dg1_mn, dg1_vr,
                                             dg2_sc, dg2_bi, dg2_mn, dg2_vr,
                                             t1, t2);

    k_conv3x3_64to2<<<(N_IMG * HW) / 256, 256, 0, stream>>>(t1, dg1_w2, d1);
    k_conv3x3_64to2<<<(N_IMG * HW) / 256, 256, 0, stream>>>(t2, dg2_w2, d2);

    k_warp_fuse<<<(N_IMG * HW) / 256, 256, 0, stream>>>(low, high, d1, d2, imap, gamma, f);

    k_outconv<<<(N_IMG * HW) / 256, 256, 0, stream>>>(f, out_w, out_b, o);
}
